// RAFT_GMA2_32968168964385
// MI455X (gfx1250) — compile-verified, matrix operand order checked by reading, not run on hardware
//
#include <hip/hip_runtime.h>
#include <math.h>

// ---------------------------------------------------------------------------
// CDNA5 (gfx1250) RAFT forward implementation.
// Matrix work runs on v_wmma_f32_16x16x32_f16:
//   - general implicit-GEMM conv (any KHxKW/stride/pad, concat views)
//   - shape-specialized 1x1-conv GEMM (immediate-offset loads, HW=4096)
//   - shape-specialized correlation GEMM (guard-free, immediate offsets)
// Each wave computes a 16(M) x 64(N) tile: one A fragment feeds 4 WMMAs.
// ---------------------------------------------------------------------------

typedef __attribute__((ext_vector_type(16))) _Float16 v16h;
typedef __attribute__((ext_vector_type(8)))  float    v8f;

__device__ inline float act_apply(int op, float v) {
    if (op == 1) return v > 0.f ? v : 0.f;            // relu
    if (op == 2) return 1.f / (1.f + __expf(-v));     // sigmoid
    if (op == 3) return tanhf(v);                     // tanh
    return v;
}

// ---------------------------------------------------------------------------
// Implicit-GEMM conv: A = weights (Cout x K), B = im2col acts (K x spatial),
// K = Cin*KH*KW.  One wave: 16 out-channels x 64 output pixels (4 subtiles).
// Fused bias + activation. Channel offset/total params implement concat views.
// A-fragment garbage (row>=Cout or k>=K) is harmless: those D rows are never
// stored, and for k>=K the matching B element is forced to 0.
// ---------------------------------------------------------------------------
__global__ __launch_bounds__(32) void conv2d_wmma_kernel(
    const float* __restrict__ x, const float* __restrict__ w,
    const float* __restrict__ bias, float* __restrict__ y,
    int N, int Cin, int Hin, int Win,
    int Cout, int Hout, int Wout,
    int KH, int KW, int stride, int padH, int padW,
    int inCtot, int inCoff, int outCtot, int outCoff, int actOp)
{
    const int lane   = threadIdx.x;
    const int laneLo = lane & 15;
    const int laneHi = (lane >> 4) & 1;
    const int m0 = blockIdx.x * 16;          // output-channel tile
    const int s0 = blockIdx.y * 64;          // spatial tile (4 x 16)
    const int K   = Cin * KH * KW;
    const int KHW = KH * KW;
    const int HWo = Hout * Wout;
    const int totalS = N * HWo;

    // A row for this lane (clamped pointer; rows >= Cout masked at store)
    const int mA = m0 + laneLo;
    const float* wrow = w + (size_t)(mA < Cout ? mA : 0) * K;

    // Per-subtile output coordinates (clamped; masked at use)
    int   nB[4], ohO[4], owO[4], sOK[4];
    const float* xb[4];
    for (int t = 0; t < 4; ++t) {
        int sB = s0 + t * 16 + laneLo;
        sOK[t] = (sB < totalS) ? 1 : 0;
        int sc = sOK[t] ? sB : 0;
        int n  = sc / HWo;
        int r  = sc % HWo;
        nB[t] = n; ohO[t] = r / Wout; owO[t] = r % Wout;
        xb[t] = x + ((size_t)n * inCtot + inCoff) * Hin * Win;
    }

    v8f acc[4] = {};

    for (int k0 = 0; k0 < K; k0 += 32) {
        // prefetch next weight chunk (global_prefetch_b8)
        __builtin_prefetch(wrow + k0 + 32, 0, 1);

        // ---- A fragment: two contiguous runs of 8 (ISA f16 layout) ----
        v16h a;
        if (k0 + 32 <= K) {
            // fast path: immediate-offset loads, no masking at all
            const float* wr = wrow + k0 + laneHi * 8;
#pragma unroll
            for (int run = 0; run < 2; ++run)
#pragma unroll
                for (int e = 0; e < 8; ++e)
                    a[run * 8 + e] = (_Float16)wr[run * 16 + e];
        } else {
            // tail chunk: clamp addresses only (values for k>=K hit B==0)
#pragma unroll
            for (int run = 0; run < 2; ++run) {
                const int kb = k0 + run * 16 + laneHi * 8;
#pragma unroll
                for (int e = 0; e < 8; ++e) {
                    int k  = kb + e;
                    int kc = (k < K) ? k : (K - 1);
                    a[run * 8 + e] = (_Float16)wrow[kc];
                }
            }
        }

        // Decompose the two run starts once (k -> ci,kh,kw), advance incrementally.
        int ci0[2], kh0[2], kw0[2];
#pragma unroll
        for (int run = 0; run < 2; ++run) {
            int kb  = k0 + run * 16 + laneHi * 8;
            int kbc = (kb < K) ? kb : 0;          // keep div results sane on tail
            int c   = kbc / KHW;
            int rem = kbc - c * KHW;
            ci0[run] = c; kh0[run] = rem / KW; kw0[run] = rem - (rem / KW) * KW;
        }

        // ---- 4 B fragments, 4 WMMAs sharing fragment A ----
#pragma unroll
        for (int t = 0; t < 4; ++t) {
            const int ohs = ohO[t] * stride - padH;
            const int ows = owO[t] * stride - padW;
            v16h b;
#pragma unroll
            for (int run = 0; run < 2; ++run) {
                const int kb = k0 + run * 16 + laneHi * 8;
                int ci = ci0[run], kh = kh0[run], kw = kw0[run];
#pragma unroll
                for (int e = 0; e < 8; ++e) {
                    int k  = kb + e;
                    int ih = ohs + kh;
                    int iw = ows + kw;
                    bool ok = (k < K) & (sOK[t] != 0) &
                              (ih >= 0) & (ih < Hin) & (iw >= 0) & (iw < Win);
                    size_t idx = ok ? (((size_t)ci * Hin + ih) * Win + iw) : 0;
                    float bv = xb[t][idx];                 // unconditional load
                    b[run * 8 + e] = (_Float16)(ok ? bv : 0.f);
                    // advance (kw,kh,ci) incrementally — no div/mod
                    ++kw;
                    if (kw == KW) { kw = 0; ++kh; if (kh == KH) { kh = 0; ++ci; } }
                }
            }
            acc[t] = __builtin_amdgcn_wmma_f32_16x16x32_f16(
                false, a, false, b, (short)0, acc[t], false, false);
        }
    }

    // ---- epilogue: bias + activation, strided concat-aware store ----
#pragma unroll
    for (int t = 0; t < 4; ++t) {
        if (!sOK[t]) continue;
#pragma unroll
        for (int r = 0; r < 8; ++r) {
            int m = m0 + r + laneHi * 8;     // C/D layout: VGPR r -> row r (+8 hi)
            if (m < Cout) {
                float v = acc[t][r] + (bias ? bias[m] : 0.f);
                v = act_apply(actOp, v);
                y[(((size_t)nB[t] * outCtot + outCoff + m) * Hout + ohO[t]) * Wout
                  + owO[t]] = v;
            }
        }
    }
}

// ---------------------------------------------------------------------------
// 1x1 stride-1 conv == dense GEMM over channels, shape-specialized on HW so
// all B loads/stores use immediate offsets from a bumped base pointer.
// Requires: Cout % 16 == 0, spatial HW % 64 == 0, grid.z = N.
// Tail chunk (K % 32 != 0): A addresses clamped, B values zero-selected
// (stray reads land in workspace guard padding).
// ---------------------------------------------------------------------------
template<int HW>
__global__ __launch_bounds__(32) void gemm1x1_wmma_kernel(
    const float* __restrict__ x, const float* __restrict__ w,
    const float* __restrict__ bias, float* __restrict__ y,
    int K, int inCtot, int inCoff, int outCtot, int outCoff, int actOp)
{
    const int lane   = threadIdx.x;
    const int laneLo = lane & 15;
    const int laneHi = (lane >> 4) & 1;
    const int m0 = blockIdx.x * 16;
    const int p0 = blockIdx.y * 64;
    const int n  = blockIdx.z;

    const float* wrow = w + (size_t)(m0 + laneLo) * K;
    const float* Bb = x + ((size_t)n * inCtot + inCoff) * HW
                    + p0 + laneLo + (size_t)laneHi * 8 * HW;

    v8f acc[4] = {};
#pragma unroll 1
    for (int k0 = 0; k0 < K; k0 += 32) {
        __builtin_prefetch(wrow + k0 + 32, 0, 1);
        v16h a;
        if (k0 + 32 <= K) {
            const float* wr = wrow + k0 + laneHi * 8;
#pragma unroll
            for (int run = 0; run < 2; ++run)
#pragma unroll
                for (int e = 0; e < 8; ++e)
                    a[run * 8 + e] = (_Float16)wr[run * 16 + e];
#pragma unroll
            for (int t = 0; t < 4; ++t) {
                v16h b;
#pragma unroll
                for (int run = 0; run < 2; ++run)
#pragma unroll
                    for (int e = 0; e < 8; ++e)
                        b[run * 8 + e] =
                            (_Float16)Bb[(size_t)(run * 16 + e) * HW + t * 16];
                acc[t] = __builtin_amdgcn_wmma_f32_16x16x32_f16(
                    false, a, false, b, (short)0, acc[t], false, false);
            }
        } else {
            // tail chunk (wave-uniform branch)
#pragma unroll
            for (int run = 0; run < 2; ++run) {
                const int kb = k0 + run * 16 + laneHi * 8;
#pragma unroll
                for (int e = 0; e < 8; ++e) {
                    int k = kb + e;
                    a[run * 8 + e] = (_Float16)wrow[(k < K) ? k : (K - 1)];
                }
            }
#pragma unroll
            for (int t = 0; t < 4; ++t) {
                v16h b;
#pragma unroll
                for (int run = 0; run < 2; ++run) {
                    const int kb = k0 + run * 16 + laneHi * 8;
#pragma unroll
                    for (int e = 0; e < 8; ++e) {
                        int k = kb + e;
                        float bv = Bb[(size_t)(run * 16 + e) * HW + t * 16];
                        b[run * 8 + e] = (_Float16)((k < K) ? bv : 0.f);
                    }
                }
                acc[t] = __builtin_amdgcn_wmma_f32_16x16x32_f16(
                    false, a, false, b, (short)0, acc[t], false, false);
            }
        }
        Bb += (size_t)32 * HW;
    }

    float* Ob = y + ((size_t)n * outCtot + outCoff + m0 + laneHi * 8) * HW
              + p0 + laneLo;
#pragma unroll
    for (int t = 0; t < 4; ++t)
#pragma unroll
        for (int r = 0; r < 8; ++r) {
            float v = acc[t][r] + bias[m0 + r + laneHi * 8];
            v = act_apply(actOp, v);
            Ob[(size_t)r * HW + t * 16] = v;
        }
}

// ---------------------------------------------------------------------------
// Correlation volume GEMM: corr[n, m, p] = scale * sum_c f1[n,c,m] * f2[n,c,p]
// Shape-specialized (C, HW compile-time): every fragment load/store is a
// global_load/store with an immediate offset from one bumped base pointer.
// HW % 64 == 0 and C % 32 == 0 -> completely guard-free.
// ---------------------------------------------------------------------------
template<int C, int HW>
__global__ __launch_bounds__(32) void corr_gemm_wmma_kernel(
    const float* __restrict__ f1, const float* __restrict__ f2,
    float* __restrict__ out, float scale)
{
    const int lane   = threadIdx.x;
    const int laneLo = lane & 15;
    const int laneHi = (lane >> 4) & 1;
    const int m0 = blockIdx.x * 16;
    const int p0 = blockIdx.y * 64;
    const int n  = blockIdx.z;

    const float* Ab = f1 + (size_t)n * C * HW + (m0 + laneLo) + (size_t)laneHi * 8 * HW;
    const float* Bb = f2 + (size_t)n * C * HW + (p0 + laneLo) + (size_t)laneHi * 8 * HW;

    v8f acc[4] = {};
#pragma unroll 1
    for (int k0 = 0; k0 < C; k0 += 32) {
        v16h a;
#pragma unroll
        for (int run = 0; run < 2; ++run)
#pragma unroll
            for (int e = 0; e < 8; ++e)
                a[run * 8 + e] = (_Float16)Ab[(size_t)(run * 16 + e) * HW];
#pragma unroll
        for (int t = 0; t < 4; ++t) {
            v16h b;
#pragma unroll
            for (int run = 0; run < 2; ++run)
#pragma unroll
                for (int e = 0; e < 8; ++e)
                    b[run * 8 + e] = (_Float16)Bb[(size_t)(run * 16 + e) * HW + t * 16];
            acc[t] = __builtin_amdgcn_wmma_f32_16x16x32_f16(
                false, a, false, b, (short)0, acc[t], false, false);
        }
        Ab += (size_t)32 * HW;
        Bb += (size_t)32 * HW;
    }

    float* Ob = out + ((size_t)n * HW + m0 + laneHi * 8) * HW + p0 + laneLo;
#pragma unroll
    for (int t = 0; t < 4; ++t)
#pragma unroll
        for (int r = 0; r < 8; ++r)
            Ob[(size_t)r * HW + t * 16] = acc[t][r] * scale;
}

// ---------------------------------------------------------------------------
// Per-(n,group) mean / rsqrt(var+eps).  blockIdx.x = n*G + g.
// ---------------------------------------------------------------------------
__global__ __launch_bounds__(256) void norm_stats_kernel(
    const float* __restrict__ x, float* __restrict__ mstd,
    int Ctot, int Coff, int G, int cpg, int HW, float eps)
{
    const int n = blockIdx.x / G, g = blockIdx.x % G;
    const float* base = x + ((size_t)n * Ctot + Coff + g * cpg) * HW;
    const int total = cpg * HW;
    float s = 0.f, ss = 0.f;
    for (int i = threadIdx.x; i < total; i += blockDim.x) {
        float v = base[i];
        s += v; ss += v * v;
    }
    __shared__ float shs[256], shq[256];
    shs[threadIdx.x] = s; shq[threadIdx.x] = ss;
    __syncthreads();
    for (int off = 128; off > 0; off >>= 1) {
        if (threadIdx.x < off) {
            shs[threadIdx.x] += shs[threadIdx.x + off];
            shq[threadIdx.x] += shq[threadIdx.x + off];
        }
        __syncthreads();
    }
    if (threadIdx.x == 0) {
        float m = shs[0] / total;
        float var = shq[0] / total - m * m;
        mstd[2 * blockIdx.x + 0] = m;
        mstd[2 * blockIdx.x + 1] = rsqrtf(var + eps);
    }
}

__global__ __launch_bounds__(256) void norm_apply_kernel(
    const float* __restrict__ x, const float* __restrict__ mstd,
    const float* __restrict__ w, const float* __restrict__ b,
    float* __restrict__ y,
    int Ctot, int Coff, int G, int cpg, int HW, int fuseRelu, long total)
{
    long i = (long)blockIdx.x * blockDim.x + threadIdx.x;
    if (i >= total) return;
    const int C = G * cpg;
    int hw = (int)(i % HW);
    long t = i / HW;
    int c = (int)(t % C);
    int n = (int)(t / C);
    int g = c / cpg;
    float m  = mstd[2 * (n * G + g) + 0];
    float rs = mstd[2 * (n * G + g) + 1];
    float v = (x[((size_t)n * Ctot + Coff + c) * HW + hw] - m) * rs;
    if (w) v = v * w[c] + b[c];
    if (fuseRelu) v = v > 0.f ? v : 0.f;
    y[((size_t)n * Ctot + Coff + c) * HW + hw] = v;
}

// Copy / unary op on a channel slice (implements concat + activations).
__global__ __launch_bounds__(256) void slice_op_kernel(
    const float* __restrict__ x, float* __restrict__ y,
    int inCtot, int inCoff, int C, int HW, int outCtot, int outCoff,
    int op, long total)
{
    long i = (long)blockIdx.x * blockDim.x + threadIdx.x;
    if (i >= total) return;
    int hw = (int)(i % HW);
    long t = i / HW;
    int c = (int)(t % C);
    int n = (int)(t / C);
    float v = x[((size_t)n * inCtot + inCoff + c) * HW + hw];
    y[((size_t)n * outCtot + outCoff + c) * HW + hw] = act_apply(op, v);
}

// y[ch off] = a * b  (contiguous a,b of shape N x C x HW)
__global__ __launch_bounds__(256) void mul_into_kernel(
    const float* __restrict__ a, const float* __restrict__ b,
    float* __restrict__ y, int C, int HW, int outCtot, int outCoff, long total)
{
    long i = (long)blockIdx.x * blockDim.x + threadIdx.x;
    if (i >= total) return;
    int hw = (int)(i % HW);
    long t = i / HW;
    int c = (int)(t % C);
    int n = (int)(t / C);
    y[((size_t)n * outCtot + outCoff + c) * HW + hw] = a[i] * b[i];
}

__global__ __launch_bounds__(256) void ew_add_kernel(
    const float* __restrict__ a, const float* __restrict__ b,
    float* __restrict__ y, long total)
{
    long i = (long)blockIdx.x * blockDim.x + threadIdx.x;
    if (i < total) y[i] = a[i] + b[i];
}

__global__ __launch_bounds__(256) void ew_sub_kernel(
    const float* __restrict__ a, const float* __restrict__ b,
    float* __restrict__ y, long total)
{
    long i = (long)blockIdx.x * blockDim.x + threadIdx.x;
    if (i < total) y[i] = a[i] - b[i];
}

__global__ __launch_bounds__(256) void gru_combine_kernel(
    float* __restrict__ h, const float* __restrict__ z,
    const float* __restrict__ q, long total)
{
    long i = (long)blockIdx.x * blockDim.x + threadIdx.x;
    if (i < total) h[i] = (1.f - z[i]) * h[i] + z[i] * q[i];
}

// x = 2*(img/255) - 1, stacking image1 (batches 0..N-1) and image2 (N..2N-1)
__global__ __launch_bounds__(256) void preprocess_kernel(
    const float* __restrict__ img1, const float* __restrict__ img2,
    float* __restrict__ y, long per, long total)
{
    long i = (long)blockIdx.x * blockDim.x + threadIdx.x;
    if (i >= total) return;
    float v = (i < per) ? img1[i] : img2[i - per];
    y[i] = 2.f * (v / 255.f) - 1.f;
}

__global__ __launch_bounds__(256) void coords_init_kernel(
    float* __restrict__ c0, float* __restrict__ c1, int H, int W, long total)
{
    long i = (long)blockIdx.x * blockDim.x + threadIdx.x;
    if (i >= total) return;
    int w = (int)(i % W);
    long t = i / W;
    int h = (int)(t % H);
    int c = (int)((t / H) % 2);
    float v = (c == 0) ? (float)w : (float)h;
    c0[i] = v; c1[i] = v;
}

__global__ __launch_bounds__(256) void avgpool2_kernel(
    const float* __restrict__ x, float* __restrict__ y,
    int h, int w, long total)
{
    long i = (long)blockIdx.x * blockDim.x + threadIdx.x;
    if (i >= total) return;
    int ow2 = w >> 1, oh2 = h >> 1;
    int ox = (int)(i % ow2);
    long t = i / ow2;
    int oy = (int)(t % oh2);
    long b = t / oh2;
    const float* p = x + ((size_t)b * h + 2 * oy) * w + 2 * ox;
    y[i] = 0.25f * (p[0] + p[1] + p[w] + p[w + 1]);
}

__device__ inline float bil_gather(const float* img, float xi, float yi,
                                   int lw, int lh)
{
    bool valid = (xi >= 0.f) && (xi <= (float)(lw - 1)) &&
                 (yi >= 0.f) && (yi <= (float)(lh - 1));
    int xc = (int)fminf(fmaxf(xi, 0.f), (float)(lw - 1));
    int yc = (int)fminf(fmaxf(yi, 0.f), (float)(lh - 1));
    float v = img[(size_t)yc * lw + xc];
    return valid ? v : 0.f;
}

// One thread per (n,h,w, offset j) for one pyramid level.
__global__ __launch_bounds__(256) void corr_lookup_kernel(
    const float* __restrict__ pyr, const float* __restrict__ coords,
    float* __restrict__ out, int N, int H, int W, int lh, int lw,
    int level, int chanBase, int Ctot, long total)
{
    long i = (long)blockIdx.x * blockDim.x + threadIdx.x;
    if (i >= total) return;
    int j = (int)(i % 81);
    long t = i / 81;
    int w = (int)(t % W);
    t /= W;
    int h = (int)(t % H);
    int n = (int)(t / H);
    float inv = 1.f / (float)(1 << level);
    float cx = coords[(((size_t)n * 2 + 0) * H + h) * W + w] * inv + (float)(j % 9 - 4);
    float cy = coords[(((size_t)n * 2 + 1) * H + h) * W + w] * inv + (float)(j / 9 - 4);
    const float* img = pyr + ((size_t)n * H * W + (size_t)h * W + w) * lh * lw;
    float x0 = floorf(cx), y0 = floorf(cy);
    float wx = cx - x0, wy = cy - y0;
    float v00 = bil_gather(img, x0,       y0,       lw, lh);
    float v01 = bil_gather(img, x0 + 1.f, y0,       lw, lh);
    float v10 = bil_gather(img, x0,       y0 + 1.f, lw, lh);
    float v11 = bil_gather(img, x0 + 1.f, y0 + 1.f, lw, lh);
    float v = v00 * (1.f - wx) * (1.f - wy) + v01 * wx * (1.f - wy)
            + v10 * (1.f - wx) * wy + v11 * wx * wy;
    out[(((size_t)n * Ctot + chanBase + j) * H + h) * W + w] = v;
}

// Convex upsampling: one thread per output pixel of (N,2,8H,8W).
__global__ __launch_bounds__(256) void upsample_flow_kernel(
    const float* __restrict__ flow, const float* __restrict__ mask,
    float* __restrict__ out, int N, int H, int W, long total)
{
    long i = (long)blockIdx.x * blockDim.x + threadIdx.x;
    if (i >= total) return;
    const int OW = 8 * W, OH = 8 * H;
    int ow = (int)(i % OW);
    long t = i / OW;
    int oh = (int)(t % OH);
    t /= OH;
    int c = (int)(t % 2);
    int n = (int)(t / 2);
    int h = oh >> 3, i8 = oh & 7;
    int w = ow >> 3, j8 = ow & 7;

    float mv[9];
    float mx = -1e30f;
#pragma unroll
    for (int k = 0; k < 9; ++k) {
        mv[k] = 0.25f * mask[(((size_t)n * 576 + k * 64 + i8 * 8 + j8) * H + h) * W + w];
        mx = fmaxf(mx, mv[k]);
    }
    float den = 0.f;
#pragma unroll
    for (int k = 0; k < 9; ++k) { mv[k] = __expf(mv[k] - mx); den += mv[k]; }
    float acc = 0.f;
#pragma unroll
    for (int k = 0; k < 9; ++k) {
        int fi = k / 3, fj = k % 3;
        int fh = h + fi - 1, fw = w + fj - 1;
        float f = (fh >= 0 && fh < H && fw >= 0 && fw < W)
                ? 8.f * flow[(((size_t)n * 2 + c) * H + fh) * W + fw] : 0.f;
        acc += mv[k] * f;
    }
    out[i] = acc / den;
}

// ---------------------------------------------------------------------------
// Host orchestration
// ---------------------------------------------------------------------------
static inline int cdiv(long a, long b) { return (int)((a + b - 1) / b); }

static void launch_conv(hipStream_t s, const float* x, const float* w,
                        const float* b, float* y,
                        int N, int Cin, int Hin, int Win,
                        int Cout, int Hout, int Wout,
                        int KH, int KW, int stride, int padH, int padW,
                        int inCtot, int inCoff, int outCtot, int outCoff, int act)
{
    dim3 grid(cdiv(Cout, 16), cdiv((long)N * Hout * Wout, 64));
    conv2d_wmma_kernel<<<grid, 32, 0, s>>>(x, w, b, y, N, Cin, Hin, Win,
        Cout, Hout, Wout, KH, KW, stride, padH, padW,
        inCtot, inCoff, outCtot, outCoff, act);
}

// 1x1 stride-1 conv at 64x64 resolution (HW=4096), Cout % 16 == 0.
static void launch_conv1x1(hipStream_t s, const float* x, const float* w,
                           const float* b, float* y, int N, int K,
                           int Cout, int inCtot, int inCoff,
                           int outCtot, int outCoff, int act)
{
    dim3 grid(Cout / 16, 4096 / 64, N);
    gemm1x1_wmma_kernel<4096><<<grid, 32, 0, s>>>(
        x, w, b, y, K, inCtot, inCoff, outCtot, outCoff, act);
}

static void launch_inorm(hipStream_t s, float* x, float* mstd,
                         int N, int C, int HW, int relu)
{
    norm_stats_kernel<<<N * C, 256, 0, s>>>(x, mstd, C, 0, C, 1, HW, 1e-5f);
    long total = (long)N * C * HW;
    norm_apply_kernel<<<cdiv(total, 256), 256, 0, s>>>(
        x, mstd, nullptr, nullptr, x, C, 0, C, 1, HW, relu, total);
}

static void launch_gnorm(hipStream_t s, float* x, const float* w, const float* b,
                         float* mstd, int N, int C, int G, int HW)
{
    int cpg = C / G;
    norm_stats_kernel<<<N * G, 256, 0, s>>>(x, mstd, C, 0, G, cpg, HW, 1e-5f);
    long total = (long)N * C * HW;
    norm_apply_kernel<<<cdiv(total, 256), 256, 0, s>>>(
        x, mstd, w, b, x, C, 0, G, cpg, HW, 0, total);
}

static void launch_slice(hipStream_t s, const float* x, float* y,
                         int N, int inCtot, int inCoff, int C, int HW,
                         int outCtot, int outCoff, int op)
{
    long total = (long)N * C * HW;
    slice_op_kernel<<<cdiv(total, 256), 256, 0, s>>>(
        x, y, inCtot, inCoff, C, HW, outCtot, outCoff, op, total);
}

// input tensor indices (jax tree-flatten: dicts in sorted-key order)
enum {
    IN_IMG1 = 0, IN_IMG2 = 1,
    F_CONV1_B = 2, F_CONV1_W = 3, F_CONV2_B = 4, F_CONV2_W = 5,
    L1B1_B1 = 6, L1B1_B2 = 7, L1B1_W1 = 8, L1B1_W2 = 9,
    L1B2_B1 = 10, L1B2_B2 = 11, L1B2_W1 = 12, L1B2_W2 = 13,
    L2B1_B1 = 14, L2B1_B2 = 15, L2B1_BD = 16, L2B1_GNB = 17, L2B1_GNW = 18,
    L2B1_W1 = 19, L2B1_W2 = 20, L2B1_WD = 21,
    L2B2_B1 = 22, L2B2_B2 = 23, L2B2_W1 = 24, L2B2_W2 = 25,
    U_CB = 26, U_CC1B = 27, U_CC1W = 28, U_CC2B = 29, U_CC2W = 30,
    U_CF1B = 31, U_CF1W = 32, U_CF2B = 33, U_CF2W = 34, U_CW = 35,
    U_FH1B = 36, U_FH1W = 37, U_FH2B = 38, U_FH2W = 39,
    U_M1B = 40, U_M1W = 41, U_M2B = 42, U_M2W = 43,
    U_Q1B = 44, U_Q1W = 45, U_Q2B = 46, U_Q2W = 47,
    U_R1B = 48, U_R1W = 49, U_R2B = 50, U_R2W = 51,
    U_Z1B = 52, U_Z1W = 53, U_Z2B = 54, U_Z2W = 55,
    IN_ITERS = 56
};

static void run_resblock(hipStream_t s, const float* X, float* OUT,
                         float* T1, float* T2, float* XD, float* mstd,
                         const float* w1, const float* b1,
                         const float* w2, const float* b2,
                         const float* wd, const float* bd,
                         const float* gnw, const float* gnb,
                         int N, int Cin, int H, int W, int Cout, int stride)
{
    const int Ho = (H + 2 - 3) / stride + 1;
    const int Wo = (W + 2 - 3) / stride + 1;
    launch_conv(s, X, w1, b1, T1, N, Cin, H, W, Cout, Ho, Wo, 3, 3, stride, 1, 1,
                Cin, 0, Cout, 0, 0);
    launch_inorm(s, T1, mstd, N, Cout, Ho * Wo, 1);
    launch_conv(s, T1, w2, b2, T2, N, Cout, Ho, Wo, Cout, Ho, Wo, 3, 3, 1, 1, 1,
                Cout, 0, Cout, 0, 0);
    launch_inorm(s, T2, mstd, N, Cout, Ho * Wo, 1);
    const float* skip = X;
    if (stride != 1) {
        launch_conv(s, X, wd, bd, XD, N, Cin, H, W, Cout, Ho, Wo, 1, 1, stride, 0, 0,
                    Cin, 0, Cout, 0, 0);
        launch_inorm(s, XD, mstd, N, Cout, Ho * Wo, 0);
        launch_gnorm(s, XD, gnw, gnb, mstd, N, Cout, Cout / 8, Ho * Wo);
        skip = XD;
    }
    long total = (long)N * Cout * Ho * Wo;
    ew_add_kernel<<<cdiv(total, 256), 256, 0, s>>>(skip, T2, OUT, total);
}

extern "C" void kernel_launch(void* const* d_in, const int* in_sizes, int n_in,
                              void* d_out, int out_size, void* d_ws, size_t ws_size,
                              hipStream_t stream)
{
    (void)in_sizes; (void)n_in; (void)out_size; (void)ws_size;
    auto P = [&](int i) { return (const float*)d_in[i]; };

    const int N = 2, B = 4;               // batch; encoder batch = 2N
    const int H4 = 64, W4 = 64, HW = H4 * W4;
    const int ITERS = 4;

    // workspace carving
    char* base = (char*)d_ws;
    auto alloc = [&](size_t nfloats) {
        float* p = (float*)base;
        base += ((nfloats * sizeof(float) + 255) & ~(size_t)255);
        return p;
    };
    float* x0    = alloc((size_t)B * 3 * 256 * 256);
    float* e1    = alloc((size_t)B * 64 * 128 * 128);
    float* e2    = alloc((size_t)B * 64 * 128 * 128);
    float* e3    = alloc((size_t)B * 64 * 128 * 128);
    float* rtA   = alloc((size_t)B * 64 * 128 * 128);   // resblock temp (max size)
    float* rtB   = alloc((size_t)B * 64 * 128 * 128);
    float* xd    = alloc((size_t)B * 128 * 64 * 64);
    float* d1    = alloc((size_t)B * 128 * 64 * 64);
    float* d2    = alloc((size_t)B * 128 * 64 * 64);
    float* fmap  = alloc((size_t)B * 256 * 64 * 64);
    float* mstd  = alloc(4096);
    float* pyr0  = alloc((size_t)N * HW * 64 * 64);
    float* pyr1  = alloc((size_t)N * HW * 32 * 32);
    float* pyr2  = alloc((size_t)N * HW * 16 * 16);
    float* pyr3  = alloc((size_t)N * HW * 8 * 8);
    float* net   = alloc((size_t)N * 128 * HW);
    float* coords0 = alloc((size_t)N * 2 * HW);
    float* coords1 = alloc((size_t)N * 2 * HW);
    float* flow    = alloc((size_t)N * 2 * HW);
    float* dflow   = alloc((size_t)N * 2 * HW);
    float* corrf = alloc((size_t)N * 324 * HW);
    float* mc1   = alloc((size_t)N * 256 * HW);
    float* cfb   = alloc((size_t)N * 256 * HW);   // concat [corr-feat(192), flow-feat(64)]
    float* xg    = alloc((size_t)N * 256 * HW);   // concat [inp(128), motion(128)]
    float* hx    = alloc((size_t)N * 384 * HW);
    float* rhx   = alloc((size_t)N * 384 * HW);
    float* zb    = alloc((size_t)N * 128 * HW);
    float* rb_   = alloc((size_t)N * 128 * HW);
    float* qb    = alloc((size_t)N * 128 * HW);
    float* fhb   = alloc((size_t)N * 256 * HW);
    float* mb1   = alloc((size_t)N * 256 * HW);
    float* maskb = alloc((size_t)N * 576 * HW);
    (void)alloc((size_t)32 * HW);   // guard pad for 1x1-GEMM tail over-reads
    float* outp  = (float*)d_out;

    // ---- preprocess: x0 = 2*(img/255)-1, stacked [image1; image2] ----
    {
        long per = (long)N * 3 * 256 * 256, total = 2 * per;
        preprocess_kernel<<<cdiv(total, 256), 256, 0, stream>>>(
            P(IN_IMG1), P(IN_IMG2), x0, per, total);
    }

    // ---- encoder ----
    launch_conv(stream, x0, P(F_CONV1_W), P(F_CONV1_B), e1,
                B, 3, 256, 256, 64, 128, 128, 7, 7, 2, 3, 3, 3, 0, 64, 0, 0);
    launch_inorm(stream, e1, mstd, B, 64, 128 * 128, 1);
    run_resblock(stream, e1, e2, rtA, rtB, xd, mstd,
                 P(L1B1_W1), P(L1B1_B1), P(L1B1_W2), P(L1B1_B2),
                 nullptr, nullptr, nullptr, nullptr, B, 64, 128, 128, 64, 1);
    run_resblock(stream, e2, e3, rtA, rtB, xd, mstd,
                 P(L1B2_W1), P(L1B2_B1), P(L1B2_W2), P(L1B2_B2),
                 nullptr, nullptr, nullptr, nullptr, B, 64, 128, 128, 64, 1);
    run_resblock(stream, e3, d1, rtA, rtB, xd, mstd,
                 P(L2B1_W1), P(L2B1_B1), P(L2B1_W2), P(L2B1_B2),
                 P(L2B1_WD), P(L2B1_BD), P(L2B1_GNW), P(L2B1_GNB),
                 B, 64, 128, 128, 128, 2);
    run_resblock(stream, d1, d2, rtA, rtB, xd, mstd,
                 P(L2B2_W1), P(L2B2_B1), P(L2B2_W2), P(L2B2_B2),
                 nullptr, nullptr, nullptr, nullptr, B, 128, 64, 64, 128, 1);
    // encoder output conv (1x1): 128 -> 256
    launch_conv1x1(stream, d2, P(F_CONV2_W), P(F_CONV2_B), fmap,
                   B, 128, 256, 128, 0, 256, 0, 0);

    const float* fmap1 = fmap;
    const float* fmap2 = fmap + (size_t)N * 256 * HW;

    // ---- correlation pyramid ----
    {
        dim3 grid(HW / 16, HW / 64, N);
        corr_gemm_wmma_kernel<256, 4096><<<grid, 32, 0, stream>>>(
            fmap1, fmap2, pyr0, 1.f / 16.f);  // 1/sqrt(256)
        long t1 = (size_t)N * HW * 32 * 32;
        avgpool2_kernel<<<cdiv(t1, 256), 256, 0, stream>>>(pyr0, pyr1, 64, 64, t1);
        long t2 = (size_t)N * HW * 16 * 16;
        avgpool2_kernel<<<cdiv(t2, 256), 256, 0, stream>>>(pyr1, pyr2, 32, 32, t2);
        long t3 = (size_t)N * HW * 8 * 8;
        avgpool2_kernel<<<cdiv(t3, 256), 256, 0, stream>>>(pyr2, pyr3, 16, 16, t3);
    }

    // ---- context: net = tanh(fmap1[:, :128]); inp = relu(fmap1[:, 128:]) into xg[:128]
    launch_slice(stream, fmap1, net, N, 256, 0,   128, HW, 128, 0, 3);
    launch_slice(stream, fmap1, xg,  N, 256, 128, 128, HW, 256, 0, 1);

    {
        long t = (long)N * 2 * HW;
        coords_init_kernel<<<cdiv(t, 256), 256, 0, stream>>>(coords0, coords1, H4, W4, t);
    }

    const float* pyr[4] = { pyr0, pyr1, pyr2, pyr3 };
    const long flowN = (long)N * 2 * HW;
    const long gruN  = (long)N * 128 * HW;

    for (int it = 0; it < ITERS; ++it) {
        // corr lookup (4 levels, radius 4)
        for (int lvl = 0; lvl < 4; ++lvl) {
            long t = (long)N * H4 * W4 * 81;
            corr_lookup_kernel<<<cdiv(t, 256), 256, 0, stream>>>(
                pyr[lvl], coords1, corrf, N, H4, W4, 64 >> lvl, 64 >> lvl,
                lvl, lvl * 81, 324, t);
        }
        ew_sub_kernel<<<cdiv(flowN, 256), 256, 0, stream>>>(coords1, coords0, flow, flowN);

        // motion encoder -> xg[:,128:256]
        launch_conv1x1(stream, corrf, P(U_CC1W), P(U_CC1B), mc1,
                       N, 324, 256, 324, 0, 256, 0, 1);
        launch_conv(stream, mc1, P(U_CC2W), P(U_CC2B), cfb,
                    N, 256, H4, W4, 192, H4, W4, 3, 3, 1, 1, 1, 256, 0, 256, 0, 1);
        launch_conv(stream, flow, P(U_CF1W), P(U_CF1B), rtA,
                    N, 2, H4, W4, 128, H4, W4, 7, 7, 1, 3, 3, 2, 0, 128, 0, 1);
        launch_conv(stream, rtA, P(U_CF2W), P(U_CF2B), cfb,
                    N, 128, H4, W4, 64, H4, W4, 3, 3, 1, 1, 1, 128, 0, 256, 192, 1);
        launch_conv(stream, cfb, P(U_CW), P(U_CB), xg,
                    N, 256, H4, W4, 126, H4, W4, 3, 3, 1, 1, 1, 256, 0, 256, 128, 1);
        launch_slice(stream, flow, xg, N, 2, 0, 2, HW, 256, 254, 0);  // motion[:,126:128]

        // GRU: x part of hx/rhx is xg (constant within the iteration)
        launch_slice(stream, xg, hx,  N, 256, 0, 256, HW, 384, 128, 0);
        launch_slice(stream, xg, rhx, N, 256, 0, 256, HW, 384, 128, 0);

        for (int pass = 0; pass < 2; ++pass) {
            const float* zw = P(pass == 0 ? U_Z1W : U_Z2W);
            const float* zb2 = P(pass == 0 ? U_Z1B : U_Z2B);
            const float* rw = P(pass == 0 ? U_R1W : U_R2W);
            const float* rb2 = P(pass == 0 ? U_R1B : U_R2B);
            const float* qw = P(pass == 0 ? U_Q1W : U_Q2W);
            const float* qb2 = P(pass == 0 ? U_Q1B : U_Q2B);
            int KHg = pass == 0 ? 1 : 5, KWg = pass == 0 ? 5 : 1;
            int pHg = pass == 0 ? 0 : 2, pWg = pass == 0 ? 2 : 0;

            launch_slice(stream, net, hx, N, 128, 0, 128, HW, 384, 0, 0);
            launch_conv(stream, hx, zw, zb2, zb, N, 384, H4, W4, 128, H4, W4,
                        KHg, KWg, 1, pHg, pWg, 384, 0, 128, 0, 2);  // sigmoid
            launch_conv(stream, hx, rw, rb2, rb_, N, 384, H4, W4, 128, H4, W4,
                        KHg, KWg, 1, pHg, pWg, 384, 0, 128, 0, 2);
            mul_into_kernel<<<cdiv(gruN, 256), 256, 0, stream>>>(
                rb_, net, rhx, 128, HW, 384, 0, gruN);
            launch_conv(stream, rhx, qw, qb2, qb, N, 384, H4, W4, 128, H4, W4,
                        KHg, KWg, 1, pHg, pWg, 384, 0, 128, 0, 3);  // tanh
            gru_combine_kernel<<<cdiv(gruN, 256), 256, 0, stream>>>(net, zb, qb, gruN);
        }

        // flow head
        launch_conv(stream, net, P(U_FH1W), P(U_FH1B), fhb,
                    N, 128, H4, W4, 256, H4, W4, 3, 3, 1, 1, 1, 128, 0, 256, 0, 1);
        launch_conv(stream, fhb, P(U_FH2W), P(U_FH2B), dflow,
                    N, 256, H4, W4, 2, H4, W4, 3, 3, 1, 1, 1, 256, 0, 2, 0, 0);
        // mask head (0.25 scale folded into the upsampler); m2 is 1x1: 256 -> 576
        launch_conv(stream, net, P(U_M1W), P(U_M1B), mb1,
                    N, 128, H4, W4, 256, H4, W4, 3, 3, 1, 1, 1, 128, 0, 256, 0, 1);
        launch_conv1x1(stream, mb1, P(U_M2W), P(U_M2B), maskb,
                       N, 256, 576, 256, 0, 576, 0, 0);

        ew_add_kernel<<<cdiv(flowN, 256), 256, 0, stream>>>(coords1, dflow, coords1, flowN);
        ew_sub_kernel<<<cdiv(flowN, 256), 256, 0, stream>>>(coords1, coords0, flow, flowN);

        long upTotal = (long)N * 2 * 512 * 512;
        upsample_flow_kernel<<<cdiv(upTotal, 256), 256, 0, stream>>>(
            flow, maskb, outp + (size_t)it * upTotal, N, H4, W4, upTotal);
    }
}
